// Block_69879117906256
// MI455X (gfx1250) — matrix operand order repaired and checked by reading, not compiled, not run on hardware
//
#include <hip/hip_runtime.h>
#include <cstdint>

// ---------------------------------------------------------------------------
// Transformer block (sigmoid attention) for MI455X / gfx1250, wave32 + WMMA.
// B=8, N=1024, C=768, H=12, D=64, HID=3072.  bf16 WMMA, fp32 accumulate.
// GEMMs: 128x128x32 block tiles, LDS double-buffered (1 barrier/K-step),
// global prefetch at distance 2.  Attention: flash-style sigmoid, scores
// never touch HBM.  Workspace requirement: ~166 MB.
// ---------------------------------------------------------------------------

typedef __attribute__((ext_vector_type(16))) __bf16 bf16x16;
typedef __attribute__((ext_vector_type(8)))  __bf16 bf16x8;
typedef __attribute__((ext_vector_type(8)))  float  f32x8;

static constexpr int Bb   = 8;
static constexpr int Nn   = 1024;
static constexpr int Cc   = 768;
static constexpr int Hh   = 12;
static constexpr int Dd   = 64;
static constexpr int HID  = 3072;
static constexpr int ROWS = Bb * Nn;          // 8192

__device__ __forceinline__ f32x8 wmma_bf16(bf16x16 a, bf16x16 b, f32x8 c) {
  return __builtin_amdgcn_wmma_f32_16x16x32_bf16(false, a, false, b, (short)0, c,
                                                 false, false);
}

// A fragment (16x32 bf16 tile, row-major in LDS, stride ld elements).
// ISA layout: lane L(<16) row M=L holds K {g*8..g*8+7, 16+g*8..16+g*8+7}, g=L>>4.
__device__ __forceinline__ bf16x16 load_a_frag(const __bf16* base, int ld, int lane) {
  int g = (lane >> 4) & 1;
  const __bf16* p = base + (lane & 15) * ld + g * 8;
  union { bf16x16 v; bf16x8 h[2]; } u;
  u.h[0] = *(const bf16x8*)(p);
  u.h[1] = *(const bf16x8*)(p + 16);
  return u.v;
}

// B fragment (32x16 bf16 tile staged as [col][k], stride ld elements).
// ISA layout: lane L holds col N=L&15, K {g*8..g*8+7, 16+g*8..16+g*8+7}, g=L>>4.
__device__ __forceinline__ bf16x16 load_b_frag(const __bf16* base, int ld, int lane) {
  const __bf16* p = base + (lane & 15) * ld + ((lane >> 4) & 1) * 8;
  union { bf16x16 v; bf16x8 h[2]; } u;
  u.h[0] = *(const bf16x8*)(p);
  u.h[1] = *(const bf16x8*)(p + 16);
  return u.v;
}

// ---------------------------------------------------------------------------
// LayerNorm over C=768, fp32 in -> bf16 out. One block per row.
// ---------------------------------------------------------------------------
__global__ __launch_bounds__(256)
void ln_kernel(const float* __restrict__ x, const float* __restrict__ w,
               const float* __restrict__ b, __bf16* __restrict__ out) {
  int row = blockIdx.x;
  const float* xr = x + (size_t)row * Cc;
  int t = threadIdx.x;
  float v0 = xr[t], v1 = xr[t + 256], v2 = xr[t + 512];
  float s  = v0 + v1 + v2;
  float s2 = v0 * v0 + v1 * v1 + v2 * v2;
  __shared__ float rs[256], rq[256];
  rs[t] = s; rq[t] = s2;
  __syncthreads();
  for (int o = 128; o > 0; o >>= 1) {
    if (t < o) { rs[t] += rs[t + o]; rq[t] += rq[t + o]; }
    __syncthreads();
  }
  float mu  = rs[0] * (1.0f / Cc);
  float var = rq[0] * (1.0f / Cc) - mu * mu;
  float inv = rsqrtf(var + 1e-5f);
  __bf16* orow = out + (size_t)row * Cc;
  orow[t]       = (__bf16)((v0 - mu) * inv * w[t]       + b[t]);
  orow[t + 256] = (__bf16)((v1 - mu) * inv * w[t + 256] + b[t + 256]);
  orow[t + 512] = (__bf16)((v2 - mu) * inv * w[t + 512] + b[t + 512]);
}

// ---------------------------------------------------------------------------
// fp32 -> bf16 elementwise (weight staging)
// ---------------------------------------------------------------------------
__global__ __launch_bounds__(256)
void cvt_kernel(const float* __restrict__ in, __bf16* __restrict__ out, int n) {
  int i = blockIdx.x * 256 + threadIdx.x;
  if (i < n) out[i] = (__bf16)in[i];
}

// ---------------------------------------------------------------------------
// Generic GEMM: out = A[M,K] @ W^T + bias, W stored row-major [Nout,K] (bf16).
// Block tile 128x128x32, 8 waves (4x2), each wave 32x64 (2x4 WMMA tiles).
// LDS double buffered: one barrier per K-step, global loads overlapped with
// WMMA, distance-2 prefetch (global_prefetch_b8).
// EPI 0: scatter bf16 into qkv[3][B*H][N][D]
// EPI 1: GELU (exact, erf) -> bf16, stride Nout
// EPI 2: fp32 out = res + (v+bias)*ls, stride Nout (== C)
// ---------------------------------------------------------------------------
template <int EPI>
__global__ __launch_bounds__(256)
void gemm_bf16(const __bf16* __restrict__ A, const __bf16* __restrict__ W,
               const float* __restrict__ bias, int K, int Nout,
               __bf16* __restrict__ outb, float* __restrict__ outf,
               const float* __restrict__ res, const float* __restrict__ ls) {
  __shared__ __align__(16) __bf16 As[2 * 128 * 40];
  __shared__ __align__(16) __bf16 Bs[2 * 128 * 40];

  int tid  = threadIdx.x;
  int wid  = tid >> 5, lane = tid & 31;
  int wm   = wid & 3, wn = wid >> 2;            // 4 x 2 wave grid
  int row0 = blockIdx.y * 128;
  int col0 = blockIdx.x * 128;

  f32x8 acc[2][4] = {};

  int r  = tid >> 2;
  int cg = (tid & 3) * 8;

  const __bf16* Arow0 = A + (size_t)(row0 + r) * K + cg;
  const __bf16* Arow1 = A + (size_t)(row0 + r + 64) * K + cg;
  const __bf16* Wrow0 = W + (size_t)(col0 + r) * K + cg;
  const __bf16* Wrow1 = W + (size_t)(col0 + r + 64) * K + cg;

  bf16x8 ra0, ra1, rb0, rb1;
  auto gload = [&](int kb) {
    ra0 = *(const bf16x8*)(Arow0 + kb);
    ra1 = *(const bf16x8*)(Arow1 + kb);
    rb0 = *(const bf16x8*)(Wrow0 + kb);
    rb1 = *(const bf16x8*)(Wrow1 + kb);
  };
  auto sstore = [&](int buf) {
    __bf16* as = As + buf * (128 * 40);
    __bf16* bs = Bs + buf * (128 * 40);
    *(bf16x8*)&as[r * 40 + cg]        = ra0;
    *(bf16x8*)&as[(r + 64) * 40 + cg] = ra1;
    *(bf16x8*)&bs[r * 40 + cg]        = rb0;
    *(bf16x8*)&bs[(r + 64) * 40 + cg] = rb1;
  };

  gload(0);
  sstore(0);
  __syncthreads();

  int cur = 0;
  for (int kb = 0; kb < K; kb += 32) {
    bool more = (kb + 32) < K;
    if (more) {
      gload(kb + 32);
      if ((kb + 64) < K) {                        // distance-2 L2 prefetch
        __builtin_prefetch(Arow0 + kb + 64, 0, 1);
        __builtin_prefetch(Wrow0 + kb + 64, 0, 1);
      }
    }

    const __bf16* as = As + cur * (128 * 40);
    const __bf16* bs = Bs + cur * (128 * 40);
    bf16x16 af[2], bw[4];
#pragma unroll
    for (int mt = 0; mt < 2; ++mt)
      af[mt] = load_a_frag(&as[(wm * 32 + mt * 16) * 40], 40, lane);
#pragma unroll
    for (int nt = 0; nt < 4; ++nt)
      bw[nt] = load_b_frag(&bs[(wn * 64 + nt * 16) * 40], 40, lane);
#pragma unroll
    for (int mt = 0; mt < 2; ++mt)
#pragma unroll
      for (int nt = 0; nt < 4; ++nt)
        acc[mt][nt] = wmma_bf16(af[mt], bw[nt], acc[mt][nt]);

    if (more) sstore(cur ^ 1);
    __syncthreads();
    cur ^= 1;
  }

#pragma unroll
  for (int mt = 0; mt < 2; ++mt)
#pragma unroll
    for (int nt = 0; nt < 4; ++nt) {
      int col = col0 + wn * 64 + nt * 16 + (lane & 15);
      float bc = bias[col];
#pragma unroll
      for (int r8 = 0; r8 < 8; ++r8) {
        int row = row0 + wm * 32 + mt * 16 + r8 + 8 * (lane >> 4);
        float v = acc[mt][nt][r8] + bc;
        if constexpr (EPI == 0) {
          int b = row >> 10, n = row & (Nn - 1);
          int which = col / Cc, rem = col - which * Cc;
          int h = rem >> 6, d = rem & 63;
          outb[(size_t)which * (Bb * Hh * Nn * Dd) +
               (((size_t)(b * Hh + h) * Nn + n) * Dd) + d] = (__bf16)v;
        } else if constexpr (EPI == 1) {
          float g = 0.5f * v * (1.0f + erff(v * 0.70710678118f));
          outb[(size_t)row * Nout + col] = (__bf16)g;
        } else {
          outf[(size_t)row * Nout + col] =
              res[(size_t)row * Nout + col] + v * ls[col];
        }
      }
    }
}

// ---------------------------------------------------------------------------
// Sigmoid attention, flash-style: per (b,h), 128 query rows per block,
// iterate 64-key tiles:  S = Q K^T * scale + bias ; P = sigmoid(S) ; O += P V.
// Scores never touch HBM.
// ---------------------------------------------------------------------------
__global__ __launch_bounds__(256)
void attn_kernel(const __bf16* __restrict__ qkv, const float* __restrict__ bias_p,
                 __bf16* __restrict__ obuf) {
  __shared__ __align__(16) __bf16 Qs[128 * 72];
  __shared__ __align__(16) __bf16 Ks[64 * 72];
  __shared__ __align__(16) __bf16 Vs[64 * 72];   // transposed: Vs[d][m]
  __shared__ __align__(16) __bf16 Ss[128 * 72];  // sigmoid scores, bf16

  const size_t HSZ = (size_t)Bb * Hh * Nn * Dd;
  int bh = blockIdx.y;                 // b*H + h
  int qb = blockIdx.x;                 // query block (128 rows)
  int tid = threadIdx.x, wid = tid >> 5, lane = tid & 31;

  const __bf16* Qg = qkv + 0 * HSZ + ((size_t)bh * Nn + qb * 128) * Dd;
  const __bf16* Kg = qkv + 1 * HSZ + (size_t)bh * Nn * Dd;
  const __bf16* Vg = qkv + 2 * HSZ + (size_t)bh * Nn * Dd;

  // Load Q tile: 128x64 = 1024 8-elem vectors, 4 per thread.
#pragma unroll
  for (int p = 0; p < 4; ++p) {
    int v = tid + p * 256;
    int rr = v >> 3, dg = (v & 7) * 8;
    *(bf16x8*)&Qs[rr * 72 + dg] = *(const bf16x8*)&Qg[(size_t)rr * Dd + dg];
  }

  float abias = bias_p[0];
  const float scale = 0.125f;          // D^-0.5
  f32x8 oacc[4] = {};

  for (int j = 0; j < Nn / 64; ++j) {
    __syncthreads();
    // Load K (row-major) and V (transposed) tiles: 64x64 each.
#pragma unroll
    for (int p = 0; p < 2; ++p) {
      int v = tid + p * 256;
      int m = v >> 3, dg = (v & 7) * 8;
      *(bf16x8*)&Ks[m * 72 + dg] =
          *(const bf16x8*)&Kg[((size_t)j * 64 + m) * Dd + dg];
      bf16x8 vv = *(const bf16x8*)&Vg[((size_t)j * 64 + m) * Dd + dg];
#pragma unroll
      for (int e = 0; e < 8; ++e) Vs[(dg + e) * 72 + m] = vv[e];
    }
    __syncthreads();

    // S = Q K^T over this wave's 16 query rows x 64 keys.
    f32x8 sacc[4] = {};
#pragma unroll
    for (int ks = 0; ks < 2; ++ks) {
      bf16x16 aq = load_a_frag(&Qs[(wid * 16) * 72 + ks * 32], 72, lane);
#pragma unroll
      for (int nt = 0; nt < 4; ++nt) {
        bf16x16 bk = load_b_frag(&Ks[(nt * 16) * 72 + ks * 32], 72, lane);
        sacc[nt] = wmma_bf16(aq, bk, sacc[nt]);
      }
    }
    // sigmoid -> bf16 into Ss (rows owned exclusively by this wave).
#pragma unroll
    for (int nt = 0; nt < 4; ++nt) {
      int colk = nt * 16 + (lane & 15);
#pragma unroll
      for (int r8 = 0; r8 < 8; ++r8) {
        int rr = wid * 16 + r8 + 8 * (lane >> 4);
        float s = sacc[nt][r8] * scale + abias;
        float p = 1.0f / (1.0f + __expf(-s));
        Ss[rr * 72 + colk] = (__bf16)p;
      }
    }
    // O += P V  (A rows from Ss belong to this wave only: no barrier needed).
#pragma unroll
    for (int ks = 0; ks < 2; ++ks) {
      bf16x16 ap = load_a_frag(&Ss[(wid * 16) * 72 + ks * 32], 72, lane);
#pragma unroll
      for (int nt = 0; nt < 4; ++nt) {
        bf16x16 bv = load_b_frag(&Vs[(nt * 16) * 72 + ks * 32], 72, lane);
        oacc[nt] = wmma_bf16(ap, bv, oacc[nt]);
      }
    }
  }

  // Write O as [B,N,C] bf16 (head-interleaved for the proj GEMM).
  int b = bh / Hh, h = bh - b * Hh;
#pragma unroll
  for (int nt = 0; nt < 4; ++nt) {
    int d = nt * 16 + (lane & 15);
#pragma unroll
    for (int r8 = 0; r8 < 8; ++r8) {
      int n = qb * 128 + wid * 16 + r8 + 8 * (lane >> 4);
      obuf[((size_t)(b * Nn + n)) * Cc + h * Dd + d] = (__bf16)oacc[nt][r8];
    }
  }
}

// ---------------------------------------------------------------------------
// Host-side orchestration
// ---------------------------------------------------------------------------
extern "C" void kernel_launch(void* const* d_in, const int* in_sizes, int n_in,
                              void* d_out, int out_size, void* d_ws, size_t ws_size,
                              hipStream_t stream) {
  const float* x      = (const float*)d_in[0];
  const float* ln1_w  = (const float*)d_in[1];
  const float* ln1_b  = (const float*)d_in[2];
  const float* qkv_w  = (const float*)d_in[3];
  const float* qkv_b  = (const float*)d_in[4];
  const float* proj_w = (const float*)d_in[5];
  const float* proj_b = (const float*)d_in[6];
  const float* a_bias = (const float*)d_in[7];
  const float* ls1    = (const float*)d_in[8];
  const float* ln2_w  = (const float*)d_in[9];
  const float* ln2_b  = (const float*)d_in[10];
  const float* w1     = (const float*)d_in[11];
  const float* b1     = (const float*)d_in[12];
  const float* w2     = (const float*)d_in[13];
  const float* b2     = (const float*)d_in[14];
  const float* ls2    = (const float*)d_in[15];
  float* out = (float*)d_out;

  // Workspace carve-up (bf16 buffers, then fp32 mid-residual).
  uint8_t* ws = (uint8_t*)d_ws;
  size_t off = 0;
  auto carve_bf16 = [&](size_t elems) {
    __bf16* p = (__bf16*)(ws + off);
    off += elems * sizeof(__bf16);
    return p;
  };
  __bf16* h1     = carve_bf16((size_t)ROWS * Cc);
  __bf16* wqkv   = carve_bf16((size_t)3 * Cc * Cc);
  __bf16* wproj  = carve_bf16((size_t)Cc * Cc);
  __bf16* w1b    = carve_bf16((size_t)HID * Cc);
  __bf16* w2b    = carve_bf16((size_t)Cc * HID);
  __bf16* qkvbuf = carve_bf16((size_t)3 * Bb * Hh * Nn * Dd);
  __bf16* obuf   = carve_bf16((size_t)ROWS * Cc);
  __bf16* h2     = carve_bf16((size_t)ROWS * Cc);
  __bf16* mbuf   = carve_bf16((size_t)ROWS * HID);
  off = (off + 255) & ~(size_t)255;
  float* xmid = (float*)(ws + off);
  off += (size_t)ROWS * Cc * sizeof(float);

  // 1. Stage weights to bf16.
  auto cvt = [&](const float* src, __bf16* dst, int n) {
    cvt_kernel<<<(n + 255) / 256, 256, 0, stream>>>(src, dst, n);
  };
  cvt(qkv_w,  wqkv,  3 * Cc * Cc);
  cvt(proj_w, wproj, Cc * Cc);
  cvt(w1,     w1b,   HID * Cc);
  cvt(w2,     w2b,   Cc * HID);

  // 2. LN1 -> h1 (bf16).
  ln_kernel<<<ROWS, 256, 0, stream>>>(x, ln1_w, ln1_b, h1);

  // 3. QKV GEMM with head-scatter epilogue.
  gemm_bf16<0><<<dim3((3 * Cc) / 128, ROWS / 128), 256, 0, stream>>>(
      h1, wqkv, qkv_b, Cc, 3 * Cc, qkvbuf, nullptr, nullptr, nullptr);

  // 4. Fused sigmoid attention.
  attn_kernel<<<dim3(Nn / 128, Bb * Hh), 256, 0, stream>>>(qkvbuf, a_bias, obuf);

  // 5. Proj GEMM + layer-scale residual -> xmid (fp32).
  gemm_bf16<2><<<dim3(Cc / 128, ROWS / 128), 256, 0, stream>>>(
      obuf, wproj, proj_b, Cc, Cc, nullptr, xmid, x, ls1);

  // 6. LN2 -> h2 (bf16).
  ln_kernel<<<ROWS, 256, 0, stream>>>(xmid, ln2_w, ln2_b, h2);

  // 7. FC1 GEMM + exact GELU -> mbuf (bf16).
  gemm_bf16<1><<<dim3(HID / 128, ROWS / 128), 256, 0, stream>>>(
      h2, w1b, b1, Cc, HID, mbuf, nullptr, nullptr, nullptr);

  // 8. FC2 GEMM + layer-scale residual -> d_out (fp32).
  gemm_bf16<2><<<dim3(Cc / 128, ROWS / 128), 256, 0, stream>>>(
      mbuf, w2b, b2, HID, Cc, nullptr, out, xmid, ls2);
}
